// GraphConvolution_39487929319800
// MI455X (gfx1250) — matrix operand order repaired and checked by reading, not compiled, not run on hardware
//
#include <hip/hip_runtime.h>
#include <hip/hip_bf16.h>

#define N_NODES_C  100000
#define N_EDGES_C  1600000
#define IN_FEAT_C  256
#define OUT_FEAT_C 128

typedef __attribute__((ext_vector_type(16))) __bf16 v16bf;
typedef __attribute__((ext_vector_type(8)))  __bf16 v8bf;
typedef __attribute__((ext_vector_type(8)))  float  v8f;

__device__ __forceinline__ unsigned short f32_to_bf16_rne(float f) {
    unsigned u = __float_as_uint(f);
    u += 0x7FFFu + ((u >> 16) & 1u);   // round-to-nearest-even
    return (unsigned short)(u >> 16);
}

// CDNA5 async DMA: global -> LDS, 16B per lane, tracked by ASYNCcnt.
// VDST VGPR carries the LDS byte offset (low 32 bits of generic shared ptr).
__device__ __forceinline__ void async_copy_b128(const unsigned short* lds_dst,
                                                const unsigned short* gsrc) {
    unsigned lds_off = (unsigned)(unsigned long long)(uintptr_t)lds_dst;
    unsigned long long ga = (unsigned long long)(uintptr_t)gsrc;
    asm volatile("global_load_async_to_lds_b128 %0, %1, off"
                 :: "v"(lds_off), "v"(ga)
                 : "memory");
}
__device__ __forceinline__ void wait_async_zero() {
    asm volatile("s_wait_asynccnt 0" ::: "memory");
}

// ---------------------------------------------------------------------------
// One-time precision conversion passes (streaming, HBM-bandwidth bound):
//   xb  = bf16(x)            [N_NODES][256]
//   Wt* = bf16(W*^T)         [128][256]  (so B panels are contiguous per col)
// ---------------------------------------------------------------------------
__global__ __launch_bounds__(256)
void cvt_x_bf16(const float* __restrict__ in, unsigned short* __restrict__ outp,
                int n4) {
    const int i = blockIdx.x * blockDim.x + threadIdx.x;
    if (i < n4) {
        const float4 f = ((const float4*)in)[i];
        ushort4 o;
        o.x = f32_to_bf16_rne(f.x);
        o.y = f32_to_bf16_rne(f.y);
        o.z = f32_to_bf16_rne(f.z);
        o.w = f32_to_bf16_rne(f.w);
        ((ushort4*)outp)[i] = o;
    }
}

__global__ __launch_bounds__(256)
void cvt_w_transpose_bf16(const float* __restrict__ W,
                          unsigned short* __restrict__ Wt) {
    const int i = blockIdx.x * blockDim.x + threadIdx.x;  // [0, 128*256)
    const int n = i >> 8;     // output row (feature col of W)
    const int k = i & 255;    // output col (K)
    Wt[n * IN_FEAT_C + k] = f32_to_bf16_rne(W[(size_t)k * OUT_FEAT_C + n]);
}

// ---------------------------------------------------------------------------
// Fused h0 = x@W0, h1 = x@W1 with bf16 WMMA, f32 accumulate.
// Block = 8 waves = 128-row M-block; x read ONCE (bf16). LDS panels are
// filled with global_load_async_to_lds_b128 (ASYNCcnt DMA) each K-step.
// ---------------------------------------------------------------------------
__global__ __launch_bounds__(256)
void gcn_gemm_fused_bf16(const unsigned short* __restrict__ xb,
                         const unsigned short* __restrict__ Wt0,
                         const unsigned short* __restrict__ Wt1,
                         float* __restrict__ h0, float* __restrict__ h1) {
    __shared__ unsigned short lA[128 * 32];   // x panel   [row][k] bf16
    __shared__ unsigned short lB0[128 * 32];  // W0^T panel [n][k]  bf16
    __shared__ unsigned short lB1[128 * 32];  // W1^T panel [n][k]  bf16

    const int tid  = threadIdx.x;
    const int lane = tid & 31;
    const int wave = tid >> 5;
    const int m0   = blockIdx.x * 128;
    const int half = lane >> 4;
    const int lm   = lane & 15;
    const bool waveValid = (m0 + wave * 16) < N_NODES_C;  // N_NODES % 16 == 0

    v8f acc0[8] = {};
    v8f acc1[8] = {};

    for (int k0 = 0; k0 < IN_FEAT_C; k0 += 32) {
        __syncthreads();  // previous iteration's LDS reads done
        // ---- async DMA staging: 3 panels x 8KB; 2x16B per thread per panel
#pragma unroll
        for (int i = 0; i < 2; ++i) {
            const int idx = tid + i * 256;       // 0..511
            const int r   = idx >> 2;            // 0..127
            const int ce  = (idx & 3) << 3;      // element offset: 0,8,16,24
            int grow = m0 + r;
            if (grow >= N_NODES_C) grow = N_NODES_C - 1;  // clamp: safe, unused
            async_copy_b128(&lA[r * 32 + ce],
                            xb + (size_t)grow * IN_FEAT_C + k0 + ce);
            async_copy_b128(&lB0[r * 32 + ce],
                            Wt0 + (size_t)r * IN_FEAT_C + k0 + ce);
            async_copy_b128(&lB1[r * 32 + ce],
                            Wt1 + (size_t)r * IN_FEAT_C + k0 + ce);
        }
        wait_async_zero();
        __syncthreads();

        if (waveValid) {
            // A (16x32 bf16): lanes 0-15: row lm, K {0..15};
            //                 lanes 16-31: row lm, K {16..31}.
            const unsigned short* arow = &lA[(wave * 16 + lm) * 32];
            const v16bf a = *(const v16bf*)(const void*)(arow + half * 16);
#pragma unroll
            for (int nt = 0; nt < 8; ++nt) {
                const unsigned short* bcol = &lB0[(nt * 16 + lm) * 32];
                const v16bf b = *(const v16bf*)(const void*)(bcol + half * 16);
                acc0[nt] = __builtin_amdgcn_wmma_f32_16x16x32_bf16(
                    false, a, false, b, (short)0, acc0[nt], false, false);
            }
#pragma unroll
            for (int nt = 0; nt < 8; ++nt) {
                const unsigned short* bcol = &lB1[(nt * 16 + lm) * 32];
                const v16bf b = *(const v16bf*)(const void*)(bcol + half * 16);
                acc1[nt] = __builtin_amdgcn_wmma_f32_16x16x32_bf16(
                    false, a, false, b, (short)0, acc1[nt], false, false);
            }
        }
    }

    if (waveValid) {
        // C/D layout: VGPR v -> M = tile + v (lanes 0-15) / tile + v + 8.
#pragma unroll
        for (int nt = 0; nt < 8; ++nt) {
#pragma unroll
            for (int v = 0; v < 8; ++v) {
                const int row = m0 + wave * 16 + v + half * 8;
                const int col = nt * 16 + lm;
                h0[(size_t)row * OUT_FEAT_C + col] = acc0[nt][v];
                h1[(size_t)row * OUT_FEAT_C + col] = acc1[nt][v];
            }
        }
    }
}

// ---------------------------------------------------------------------------
// out += A_coo @ h.  rows sorted -> accumulate runs in registers, flush with
// global f32 atomics only on row change. 1 wave = 32 edges; lane owns 4 feats.
// ---------------------------------------------------------------------------
__global__ __launch_bounds__(256)
void gcn_spmm_atomic(const int* __restrict__ rows, const int* __restrict__ cols,
                     const float* __restrict__ vals, const float* __restrict__ h,
                     float* __restrict__ out) {
    const int gwave = (int)((blockIdx.x * blockDim.x + threadIdx.x) >> 5);
    const int lane  = (int)(threadIdx.x & 31);
    const long e0 = (long)gwave * 32;
    if (e0 >= (long)N_EDGES_C) return;
    const int cnt = (int)(((long)N_EDGES_C - e0) < 32 ? ((long)N_EDGES_C - e0) : 32);

    int myRow = 0, myCol = 0;
    float myVal = 0.f;
    if (lane < cnt) {
        myRow = rows[e0 + lane];
        myCol = cols[e0 + lane];
        myVal = vals[e0 + lane];
    }

    const int fo = lane << 2;
    float4 acc = make_float4(0.f, 0.f, 0.f, 0.f);
    int curRow = __shfl(myRow, 0, 32);

    for (int i = 0; i < cnt; ++i) {
        const int   r = __shfl(myRow, i, 32);
        const int   c = __shfl(myCol, i, 32);
        const float v = __shfl(myVal, i, 32);
        if (r != curRow) {  // uniform branch
            float* o = out + (size_t)curRow * OUT_FEAT_C + fo;
            atomicAdd(o + 0, acc.x);
            atomicAdd(o + 1, acc.y);
            atomicAdd(o + 2, acc.z);
            atomicAdd(o + 3, acc.w);
            acc = make_float4(0.f, 0.f, 0.f, 0.f);
            curRow = r;
        }
        const float4 g = *(const float4*)(h + (size_t)c * OUT_FEAT_C + fo);
        acc.x = fmaf(v, g.x, acc.x);
        acc.y = fmaf(v, g.y, acc.y);
        acc.z = fmaf(v, g.z, acc.z);
        acc.w = fmaf(v, g.w, acc.w);
    }
    float* o = out + (size_t)curRow * OUT_FEAT_C + fo;
    atomicAdd(o + 0, acc.x);
    atomicAdd(o + 1, acc.y);
    atomicAdd(o + 2, acc.z);
    atomicAdd(o + 3, acc.w);
}

__global__ __launch_bounds__(256)
void gcn_relu(float* __restrict__ out, int n4) {
    const int i = blockIdx.x * blockDim.x + threadIdx.x;
    if (i < n4) {
        float4 v = ((float4*)out)[i];
        v.x = fmaxf(v.x, 0.f);
        v.y = fmaxf(v.y, 0.f);
        v.z = fmaxf(v.z, 0.f);
        v.w = fmaxf(v.w, 0.f);
        ((float4*)out)[i] = v;
    }
}

extern "C" void kernel_launch(void* const* d_in, const int* in_sizes, int n_in,
                              void* d_out, int out_size, void* d_ws, size_t ws_size,
                              hipStream_t stream) {
    const float* x     = (const float*)d_in[0];
    const int*   rows0 = (const int*)d_in[1];
    const int*   cols0 = (const int*)d_in[2];
    const float* vals0 = (const float*)d_in[3];
    const int*   rows1 = (const int*)d_in[4];
    const int*   cols1 = (const int*)d_in[5];
    const float* vals1 = (const float*)d_in[6];
    const float* W0    = (const float*)d_in[7];
    const float* W1    = (const float*)d_in[8];
    float* out = (float*)d_out;

    // Workspace layout (bytes, all 16B aligned):
    //   h0:  51.2 MB | h1: 51.2 MB | xb(bf16 x): 51.2 MB | Wt0/Wt1: 64KB each
    char* ws = (char*)d_ws;
    const size_t hBytes  = (size_t)N_NODES_C * OUT_FEAT_C * sizeof(float);
    const size_t xbBytes = (size_t)N_NODES_C * IN_FEAT_C * sizeof(unsigned short);
    float*          h0  = (float*)(ws);
    float*          h1  = (float*)(ws + hBytes);
    unsigned short* xb  = (unsigned short*)(ws + 2 * hBytes);
    unsigned short* Wt0 = (unsigned short*)(ws + 2 * hBytes + xbBytes);
    unsigned short* Wt1 = Wt0 + (size_t)OUT_FEAT_C * IN_FEAT_C;

    hipMemsetAsync(out, 0, hBytes, stream);

    // bf16 conversion passes
    const int xN4 = N_NODES_C * IN_FEAT_C / 4;
    cvt_x_bf16<<<(xN4 + 255) / 256, 256, 0, stream>>>(x, xb, xN4);
    const int wN = OUT_FEAT_C * IN_FEAT_C;  // 32768
    cvt_w_transpose_bf16<<<wN / 256, 256, 0, stream>>>(W0, Wt0);
    cvt_w_transpose_bf16<<<wN / 256, 256, 0, stream>>>(W1, Wt1);

    // fused WMMA GEMM: h0, h1
    const int gemmBlocks = (N_NODES_C + 127) / 128;
    gcn_gemm_fused_bf16<<<gemmBlocks, 256, 0, stream>>>(xb, Wt0, Wt1, h0, h1);

    // sparse aggregation
    const int nWaves     = (N_EDGES_C + 31) / 32;
    const int spmmBlocks = (nWaves + 7) / 8;
    gcn_spmm_atomic<<<spmmBlocks, 256, 0, stream>>>(rows0, cols0, vals0, h0, out);
    gcn_spmm_atomic<<<spmmBlocks, 256, 0, stream>>>(rows1, cols1, vals1, h1, out);

    const int n4 = N_NODES_C * OUT_FEAT_C / 4;
    gcn_relu<<<(n4 + 255) / 256, 256, 0, stream>>>(out, n4);
}
